// MaceNet_15934328669059
// MI455X (gfx1250) — matrix operand order repaired and checked by reading, not compiled, not run on hardware
//
#include <hip/hip_runtime.h>

typedef _Float16 half8  __attribute__((ext_vector_type(8)));
typedef _Float16 half16 __attribute__((ext_vector_type(16)));
typedef float    float8 __attribute__((ext_vector_type(8)));

constexpr int N_NODES = 768;
constexpr int F_FEAT  = 128;
constexpr int V_FEAT  = 64;
constexpr int R_HID   = 16;
constexpr float INV_AVG = 1.0f / 64.0f;
constexpr int SEGS      = 8;
constexpr int S_PER_SEG = N_NODES / SEGS;  // 96
constexpr int R_TILES   = N_NODES / 16;    // 48

union V16U { half16 v; half8 h[2]; };

// Fast silu: v_exp + v_rcp + v_mul (result feeds f16 WMMA operands; ~1 ulp rcp is plenty).
__device__ __forceinline__ float fast_silu(float z) {
  return z * __builtin_amdgcn_rcpf(1.0f + __expf(-z));
}

// ---------------------------------------------------------------- init
__global__ __launch_bounds__(128) void init_kernel(const float* __restrict__ embed,
                                                   float* __restrict__ hs,
                                                   float* __restrict__ hv) {
  int n = blockIdx.x, t = threadIdx.x;
  hs[n * F_FEAT + t] = embed[t];
  hv[n * V_FEAT * 3 + t] = 0.0f;
  if (t < 64) hv[n * V_FEAT * 3 + 128 + t] = 0.0f;
}

// ---------------------------------------------------------------- per-layer prep
// hsT (f16, [F][N]), P = hs@Wsv, PT (f16, [V][N]); zero the aggregation buffers.
__global__ __launch_bounds__(128) void prep_kernel(const float* __restrict__ hs,
                                                   const float* __restrict__ Wsv,
                                                   float* __restrict__ P,
                                                   _Float16* __restrict__ hsT,
                                                   _Float16* __restrict__ PT,
                                                   float* __restrict__ agg_s,
                                                   float* __restrict__ agg_v) {
  int n = blockIdx.x, t = threadIdx.x;
  __shared__ float row[F_FEAT];
  float h = hs[n * F_FEAT + t];
  row[t] = h;
  hsT[t * N_NODES + n] = (_Float16)h;
  agg_s[n * F_FEAT + t] = 0.0f;
  agg_v[n * V_FEAT * 3 + t] = 0.0f;
  if (t < 64) agg_v[n * V_FEAT * 3 + 128 + t] = 0.0f;
  __syncthreads();
  if (t < V_FEAT) {
    float a = 0.0f;
#pragma unroll 8
    for (int f = 0; f < F_FEAT; ++f) a += row[f] * Wsv[f * V_FEAT + t];
    P[n * V_FEAT + t] = a;
    PT[t * N_NODES + n] = (_Float16)a;
  }
}

// ---------------------------------------------------------------- scalar-path edge GEMM
// agg_s[r,f] += (1/64) * sum_h Wr2s[h,f] * sum_s silu(d_rs*Wr1[h]) * hs[s,f]
// 8 waves per block; wave w owns f-slice [16w,16w+16); 16 h-accumulators each.
__global__ __launch_bounds__(256) void edge_scalar_kernel(const float* __restrict__ x,
                                                          const float* __restrict__ w1,
                                                          const float* __restrict__ w2s,
                                                          const _Float16* __restrict__ hsT,
                                                          float* __restrict__ agg_s) {
  const int rblk = blockIdx.x % R_TILES;
  const int seg  = blockIdx.x / R_TILES;
  const int r0 = rblk * 16;
  const int tid = threadIdx.x;
  const int lane = tid & 31;
  const int wid = tid >> 5;
  const int f0 = wid * 16;
  const int nlo = lane & 15;
  const int hi16 = (lane >= 16);
  __shared__ _Float16 At[R_HID][16][40];  // padded (40 halves) to spread LDS banks

  float w1r[R_HID];
#pragma unroll
  for (int h = 0; h < R_HID; ++h) w1r[h] = w1[h];

  float8 acc[R_HID];
#pragma unroll
  for (int h = 0; h < R_HID; ++h) acc[h] = (float8)0.0f;

  for (int s0 = seg * S_PER_SEG; s0 < seg * S_PER_SEG + S_PER_SEG; s0 += 32) {
    // Build the 16 radial A-tiles (16 r x 32 s) in LDS: 512 pairs, 2 per thread.
#pragma unroll
    for (int p = 0; p < 2; ++p) {
      int pr = tid * 2 + p;
      int sl = pr & 31, rl = pr >> 5;
      int sg = s0 + sl, rg = r0 + rl;
      float dx = x[sg * 3 + 0] - x[rg * 3 + 0];
      float dy = x[sg * 3 + 1] - x[rg * 3 + 1];
      float dz = x[sg * 3 + 2] - x[rg * 3 + 2];
      float dist = sqrtf(dx * dx + dy * dy + dz * dz);  // == 0 on diagonal -> silu = 0
#pragma unroll
      for (int h = 0; h < R_HID; ++h)
        At[h][rl][sl] = (_Float16)fast_silu(dist * w1r[h]);
    }
    __syncthreads();
    // B tile (32x16 f16): lane = column f0+nlo; lanes 0-15 K 0-7 & 16-23; lanes 16-31 K 8-15 & 24-31.
    V16U b;
    const _Float16* bp = hsT + (f0 + nlo) * N_NODES + s0 + (hi16 ? 8 : 0);
    b.h[0] = *(const half8*)(bp);
    b.h[1] = *(const half8*)(bp + 16);
#pragma unroll
    for (int h = 0; h < R_HID; ++h) {
      V16U a;  // A layout: lanes 0-15 K 0-7 & 16-23; lanes 16-31 K 8-15 & 24-31
      const _Float16* ap = &At[h][nlo][hi16 ? 8 : 0];
      a.h[0] = *(const half8*)(ap);
      a.h[1] = *(const half8*)(ap + 16);
      acc[h] = __builtin_amdgcn_wmma_f32_16x16x32_f16(false, a.v, false, b.v,
                                                      (short)0, acc[h], false, false);
    }
    __syncthreads();
  }
  // Contract over h with Wr2s and accumulate to global.
  float wc[R_HID];
#pragma unroll
  for (int h = 0; h < R_HID; ++h) wc[h] = w2s[h * F_FEAT + f0 + nlo];
#pragma unroll
  for (int i = 0; i < 8; ++i) {
    float v = 0.0f;
#pragma unroll
    for (int h = 0; h < R_HID; ++h) v += acc[h][i] * wc[h];
    int M = i + (hi16 ? 8 : 0);
    atomicAdd(&agg_s[(r0 + M) * F_FEAT + f0 + nlo], v * INV_AVG);
  }
}

// ---------------------------------------------------------------- vector-path edge GEMM
// agg_v[r,v,d] += (1/64)*sum_h Wr2v[h,v]*sum_s silu(d_rs*Wr1[h])*u_d[r,s]*P[s,v]
// 12 waves: wave -> (d = wid/4, v-slice = wid%4); 16 h-accumulators per wave.
__global__ __launch_bounds__(384) void edge_vector_kernel(const float* __restrict__ x,
                                                          const float* __restrict__ w1,
                                                          const float* __restrict__ w2v,
                                                          const _Float16* __restrict__ PT,
                                                          float* __restrict__ agg_v) {
  const int rblk = blockIdx.x % R_TILES;
  const int seg  = blockIdx.x / R_TILES;
  const int r0 = rblk * 16;
  const int tid = threadIdx.x;
  const int lane = tid & 31;
  const int wid = tid >> 5;      // 0..11
  const int dW = wid >> 2;       // 0..2  (spatial component)
  const int v0 = (wid & 3) * 16; // vector-channel slice
  const int nlo = lane & 15;
  const int hi16 = (lane >= 16);
  __shared__ _Float16 At[R_HID][3][16][40];  // 61,440 B

  float w1r[R_HID];
#pragma unroll
  for (int h = 0; h < R_HID; ++h) w1r[h] = w1[h];

  float8 acc[R_HID];
#pragma unroll
  for (int h = 0; h < R_HID; ++h) acc[h] = (float8)0.0f;

  for (int s0 = seg * S_PER_SEG; s0 < seg * S_PER_SEG + S_PER_SEG; s0 += 32) {
    for (int pr = tid; pr < 512; pr += 384) {
      int sl = pr & 31, rl = pr >> 5;
      int sg = s0 + sl, rg = r0 + rl;
      float dx = x[sg * 3 + 0] - x[rg * 3 + 0];
      float dy = x[sg * 3 + 1] - x[rg * 3 + 1];
      float dz = x[sg * 3 + 2] - x[rg * 3 + 2];
      float dist = sqrtf(dx * dx + dy * dy + dz * dz);
      float rin = __builtin_amdgcn_rcpf(dist + 1e-8f);
      float ux = dx * rin, uy = dy * rin, uz = dz * rin;  // = 0 on diagonal
#pragma unroll
      for (int h = 0; h < R_HID; ++h) {
        float a = fast_silu(dist * w1r[h]);
        At[h][0][rl][sl] = (_Float16)(a * ux);
        At[h][1][rl][sl] = (_Float16)(a * uy);
        At[h][2][rl][sl] = (_Float16)(a * uz);
      }
    }
    __syncthreads();
    V16U b;
    const _Float16* bp = PT + (v0 + nlo) * N_NODES + s0 + (hi16 ? 8 : 0);
    b.h[0] = *(const half8*)(bp);
    b.h[1] = *(const half8*)(bp + 16);
#pragma unroll
    for (int h = 0; h < R_HID; ++h) {
      V16U a;
      const _Float16* ap = &At[h][dW][nlo][hi16 ? 8 : 0];
      a.h[0] = *(const half8*)(ap);
      a.h[1] = *(const half8*)(ap + 16);
      acc[h] = __builtin_amdgcn_wmma_f32_16x16x32_f16(false, a.v, false, b.v,
                                                      (short)0, acc[h], false, false);
    }
    __syncthreads();
  }
  float wc[R_HID];
#pragma unroll
  for (int h = 0; h < R_HID; ++h) wc[h] = w2v[h * V_FEAT + v0 + nlo];
#pragma unroll
  for (int i = 0; i < 8; ++i) {
    float v = 0.0f;
#pragma unroll
    for (int h = 0; h < R_HID; ++h) v += acc[h][i] * wc[h];
    int M = i + (hi16 ? 8 : 0);
    atomicAdd(&agg_v[((r0 + M) * V_FEAT + v0 + nlo) * 3 + dW], v * INV_AVG);
  }
}

// ---------------------------------------------------------------- node update (exact fp32)
__global__ __launch_bounds__(256) void update_kernel(const float* __restrict__ agg_s,
                                                     const float* __restrict__ agg_v,
                                                     const float* __restrict__ Wss,
                                                     const float* __restrict__ Wvs,
                                                     const float* __restrict__ Wvv,
                                                     float* __restrict__ hs,
                                                     float* __restrict__ hv) {
  int n = blockIdx.x, t = threadIdx.x;
  __shared__ float s_aggs[F_FEAT];
  __shared__ float s_inv[V_FEAT];
  __shared__ float s_aggv[V_FEAT * 3];
  if (t < 128) {
    s_aggs[t] = agg_s[n * F_FEAT + t];
  } else if (t < 192) {
    int v = t - 128;
    float a0 = agg_v[(n * V_FEAT + v) * 3 + 0];
    float a1 = agg_v[(n * V_FEAT + v) * 3 + 1];
    float a2 = agg_v[(n * V_FEAT + v) * 3 + 2];
    s_aggv[v * 3 + 0] = a0; s_aggv[v * 3 + 1] = a1; s_aggv[v * 3 + 2] = a2;
    s_inv[v] = a0 * a0 + a1 * a1 + a2 * a2;
  }
  __syncthreads();
  if (t < 128) {
    float acc = hs[n * F_FEAT + t];
#pragma unroll 8
    for (int k = 0; k < 128; ++k) acc += s_aggs[k] * Wss[k * F_FEAT + t];
#pragma unroll 8
    for (int v = 0; v < 64; ++v) acc += s_inv[v] * Wvs[v * F_FEAT + t];
    hs[n * F_FEAT + t] = acc / (1.0f + __expf(-acc));  // exact silu for state update
  }
  if (t < 192) {
    int w_ = t / 3, d = t % 3;
    float acc = hv[(n * V_FEAT + w_) * 3 + d];
#pragma unroll 8
    for (int v = 0; v < 64; ++v) acc += s_aggv[v * 3 + d] * Wvv[v * V_FEAT + w_];
    hv[(n * V_FEAT + w_) * 3 + d] = acc;
  }
}

// ---------------------------------------------------------------- readout (exact fp32)
__global__ __launch_bounds__(384) void readout_kernel(const float* __restrict__ hs,
                                                      const float* __restrict__ hv,
                                                      const float* __restrict__ Ws1,
                                                      const float* __restrict__ Ws2,
                                                      const float* __restrict__ Wv1,
                                                      const float* __restrict__ Wv2,
                                                      float* __restrict__ out) {
  int n = blockIdx.x, t = threadIdx.x;
  __shared__ float s_hs[128], s_hv[192], s_t1[128], s_t2[384];
  if (t < 128) s_hs[t] = hs[n * 128 + t];
  else if (t < 320) s_hv[t - 128] = hv[n * 192 + (t - 128)];
  __syncthreads();
  if (t < 128) {
    float a = 0.0f;
#pragma unroll 8
    for (int f = 0; f < 128; ++f) a += s_hs[f] * Ws1[f * 128 + t];
    s_t1[t] = a / (1.0f + __expf(-a));
  }
  {
    int d = t / 128, w_ = t % 128;
    float a = 0.0f;
#pragma unroll 8
    for (int v = 0; v < 64; ++v) a += s_hv[v * 3 + d] * Wv1[v * 128 + w_];
    s_t2[t] = a;
  }
  __syncthreads();
  if (t < 64) {
    float a = 0.0f;
#pragma unroll 8
    for (int k = 0; k < 128; ++k) a += s_t1[k] * Ws2[k * 64 + t];
    out[n * 160 + t] = a;
  } else if (t < 160) {
    int q = t - 64, wo = q / 3, d = q % 3;
    float a = 0.0f;
#pragma unroll 8
    for (int w_ = 0; w_ < 128; ++w_) a += s_t2[d * 128 + w_] * Wv2[w_ * 32 + wo];
    out[n * 160 + 64 + wo * 3 + d] = a;
  }
}

// ---------------------------------------------------------------- launch
extern "C" void kernel_launch(void* const* d_in, const int* in_sizes, int n_in,
                              void* d_out, int out_size, void* d_ws, size_t ws_size,
                              hipStream_t stream) {
  (void)in_sizes; (void)n_in; (void)out_size; (void)ws_size;
  const float* x      = (const float*)d_in[0];
  // d_in[1] senders, d_in[2] receivers: unused (dense fully-connected restructure)
  const float* embed  = (const float*)d_in[3];
  const float* Wr1    = (const float*)d_in[4];
  const float* Wr2s   = (const float*)d_in[5];
  const float* Wr2v   = (const float*)d_in[6];
  const float* Wsv    = (const float*)d_in[7];
  const float* Wss    = (const float*)d_in[8];
  const float* Wvs    = (const float*)d_in[9];
  const float* Wvv    = (const float*)d_in[10];
  const float* Wro_s1 = (const float*)d_in[11];
  const float* Wro_s2 = (const float*)d_in[12];
  const float* Wro_v1 = (const float*)d_in[13];
  const float* Wro_v2 = (const float*)d_in[14];
  float* out = (float*)d_out;

  char* w = (char*)d_ws;
  float* hs    = (float*)w; w += (size_t)N_NODES * F_FEAT * 4;
  float* hv    = (float*)w; w += (size_t)N_NODES * V_FEAT * 3 * 4;
  float* agg_s = (float*)w; w += (size_t)N_NODES * F_FEAT * 4;
  float* agg_v = (float*)w; w += (size_t)N_NODES * V_FEAT * 3 * 4;
  float* P     = (float*)w; w += (size_t)N_NODES * V_FEAT * 4;
  _Float16* hsT = (_Float16*)w; w += (size_t)F_FEAT * N_NODES * 2;
  _Float16* PT  = (_Float16*)w; w += (size_t)V_FEAT * N_NODES * 2;

  init_kernel<<<N_NODES, 128, 0, stream>>>(embed, hs, hv);
  for (int l = 0; l < 2; ++l) {
    prep_kernel<<<N_NODES, 128, 0, stream>>>(hs, Wsv + l * F_FEAT * V_FEAT, P, hsT, PT,
                                             agg_s, agg_v);
    edge_scalar_kernel<<<R_TILES * SEGS, 256, 0, stream>>>(
        x, Wr1 + l * R_HID, Wr2s + l * R_HID * F_FEAT, hsT, agg_s);
    edge_vector_kernel<<<R_TILES * SEGS, 384, 0, stream>>>(
        x, Wr1 + l * R_HID, Wr2v + l * R_HID * V_FEAT, PT, agg_v);
    update_kernel<<<N_NODES, 256, 0, stream>>>(agg_s, agg_v, Wss + l * F_FEAT * F_FEAT,
                                               Wvs + l * V_FEAT * F_FEAT,
                                               Wvv + l * V_FEAT * V_FEAT, hs, hv);
  }
  readout_kernel<<<N_NODES, 384, 0, stream>>>(hs, hv, Wro_s1, Wro_s2, Wro_v1, Wro_v2, out);
}
